// VectorQuantizer_43224550867214
// MI455X (gfx1250) — matrix operand order repaired and checked by reading, not compiled, not run on hardware
//
#include <hip/hip_runtime.h>
#include <hip/hip_bf16.h>
#include <stdint.h>

typedef __attribute__((ext_vector_type(16))) __bf16 v16bf;
typedef __attribute__((ext_vector_type(8)))  float  v8f;

#define NUM_CODES 8192
#define CODE_DIM  512
#define NROWS     16384        // 16*32*32
#define MT        64           // rows per block in argmin kernel
#define LDS_ROW   520          // padded row stride in ushorts (bank-conflict-free)

static __device__ __forceinline__ unsigned short f2bf(float x) {
  union { float f; unsigned int u; } v; v.f = x;
  unsigned int r = v.u + 0x7FFFu + ((v.u >> 16) & 1u);   // round-to-nearest-even
  return (unsigned short)(r >> 16);
}
static __device__ __forceinline__ float bf2f(unsigned short h) {
  union { unsigned int u; float f; } v; v.u = ((unsigned int)h) << 16;
  return v.f;
}

// ---------------------------------------------------------------------------
// Zero histogram
// ---------------------------------------------------------------------------
__global__ void vq_zero(int* __restrict__ hist) {
  int i = blockIdx.x * 256 + threadIdx.x;
  if (i < NUM_CODES) hist[i] = 0;
}

// ---------------------------------------------------------------------------
// Convert codebook -> bf16 hi/lo planes, compute |c|^2 per code
// ---------------------------------------------------------------------------
__global__ void vq_prep_codebook(const float* __restrict__ cb,
                                 unsigned short* __restrict__ chi,
                                 unsigned short* __restrict__ clo,
                                 float* __restrict__ c2) {
  __shared__ float red[256];
  const int code = blockIdx.x;
  const float* src = cb + (size_t)code * CODE_DIM;
  unsigned short* dh = chi + (size_t)code * CODE_DIM;
  unsigned short* dl = clo + (size_t)code * CODE_DIM;
  float s = 0.f;
  for (int e = threadIdx.x; e < CODE_DIM; e += 256) {
    float x = src[e];
    unsigned short hb = f2bf(x);
    dh[e] = hb;
    dl[e] = f2bf(x - bf2f(hb));
    s += x * x;
  }
  red[threadIdx.x] = s;
  __syncthreads();
  for (int off = 128; off > 0; off >>= 1) {
    if (threadIdx.x < off) red[threadIdx.x] += red[threadIdx.x + off];
    __syncthreads();
  }
  if (threadIdx.x == 0) c2[code] = red[0];
}

// ---------------------------------------------------------------------------
// WMMA argmin kernel: 64 rows/block, 8 waves each scan 1024 codes.
// Two 16-code chunks per iteration share the same A fragments
// (6 WMMAs per A-fragment load pair). Distance surrogate s = |c|^2 - 2 x.c.
// ---------------------------------------------------------------------------
__launch_bounds__(256)
__global__ void vq_argmin(const float* __restrict__ z_e,
                          const unsigned short* __restrict__ chi,
                          const unsigned short* __restrict__ clo,
                          const float* __restrict__ c2,
                          int* __restrict__ idx_out,
                          float* __restrict__ idxf_out) {
  extern __shared__ char smem[];
  unsigned short* zhi = (unsigned short*)smem;            // MT*LDS_ROW
  unsigned short* zlo = zhi + MT * LDS_ROW;               // MT*LDS_ROW
  float* redd = (float*)(zlo + MT * LDS_ROW);             // 8*MT
  int*   redi = (int*)(redd + 8 * MT);                    // 8*MT

  const int tid = threadIdx.x;
  const int row_base = blockIdx.x * MT;

  // Stage z tile, split fp32 -> bf16 hi + lo in LDS
  for (int i = tid; i < MT * CODE_DIM; i += 256) {
    float x = z_e[(size_t)row_base * CODE_DIM + i];
    int row = i >> 9, k = i & 511;
    unsigned short hb = f2bf(x);
    zhi[row * LDS_ROW + k] = hb;
    zlo[row * LDS_ROW + k] = f2bf(x - bf2f(hb));
  }
  __syncthreads();

  const int wave = tid >> 5;
  const int lane = tid & 31;
  const int col  = lane & 15;     // code column within 16-wide chunk / C-frag N
  const int hf   = lane >> 4;     // which half of the K-split this lane owns

  const v8f vzero = {0.f, 0.f, 0.f, 0.f, 0.f, 0.f, 0.f, 0.f};

  v8f mind[4];
  int mini[4][8];
#pragma unroll
  for (int s = 0; s < 4; ++s) {
#pragma unroll
    for (int j = 0; j < 8; ++j) { mind[s][j] = 3.4e38f; mini[s][j] = 0; }
  }

  const int code0 = wave * (NUM_CODES / 8);

  for (int cc = 0; cc < (NUM_CODES / 8) / 32; ++cc) {
    const int codeA = code0 + cc * 32 + col;        // first 16-code chunk
    const int codeB = codeA + 16;                   // second 16-code chunk
    const unsigned short* bh0 = chi + (size_t)codeA * CODE_DIM + hf * 16;
    const unsigned short* bl0 = clo + (size_t)codeA * CODE_DIM + hf * 16;
    const unsigned short* bh1 = chi + (size_t)codeB * CODE_DIM + hf * 16;
    const unsigned short* bl1 = clo + (size_t)codeB * CODE_DIM + hf * 16;

    v8f acc0[4], acc1[4];
#pragma unroll
    for (int s = 0; s < 4; ++s) { acc0[s] = vzero; acc1[s] = vzero; }

    for (int kk = 0; kk < 16; ++kk) {
      // B fragments: 16 contiguous bf16 of each lane's code row (k = hf*16 + s, same order as A)
      union { uint4 q[2]; v16bf v; } Bh0, Bl0, Bh1, Bl1;
      Bh0.q[0] = *(const uint4*)(bh0 + kk * 32);
      Bh0.q[1] = *(const uint4*)(bh0 + kk * 32 + 8);
      Bl0.q[0] = *(const uint4*)(bl0 + kk * 32);
      Bl0.q[1] = *(const uint4*)(bl0 + kk * 32 + 8);
      Bh1.q[0] = *(const uint4*)(bh1 + kk * 32);
      Bh1.q[1] = *(const uint4*)(bh1 + kk * 32 + 8);
      Bl1.q[0] = *(const uint4*)(bl1 + kk * 32);
      Bl1.q[1] = *(const uint4*)(bl1 + kk * 32 + 8);
#pragma unroll
      for (int s = 0; s < 4; ++s) {
        // A fragment: row = lane%16 of subtile, K = hf*16..hf*16+15 contiguous
        const unsigned short* ah = zhi + (s * 16 + col) * LDS_ROW + kk * 32 + hf * 16;
        const unsigned short* al = zlo + (s * 16 + col) * LDS_ROW + kk * 32 + hf * 16;
        union { uint4 q[2]; v16bf v; } Ah, Al;
        Ah.q[0] = *(const uint4*)(ah);
        Ah.q[1] = *(const uint4*)(ah + 8);
        Al.q[0] = *(const uint4*)(al);
        Al.q[1] = *(const uint4*)(al + 8);
        acc0[s] = __builtin_amdgcn_wmma_f32_16x16x32_bf16(false, Ah.v, false, Bh0.v,
                                                          (short)0, acc0[s], false, false);
        acc1[s] = __builtin_amdgcn_wmma_f32_16x16x32_bf16(false, Ah.v, false, Bh1.v,
                                                          (short)0, acc1[s], false, false);
        acc0[s] = __builtin_amdgcn_wmma_f32_16x16x32_bf16(false, Ah.v, false, Bl0.v,
                                                          (short)0, acc0[s], false, false);
        acc1[s] = __builtin_amdgcn_wmma_f32_16x16x32_bf16(false, Ah.v, false, Bl1.v,
                                                          (short)0, acc1[s], false, false);
        acc0[s] = __builtin_amdgcn_wmma_f32_16x16x32_bf16(false, Al.v, false, Bh0.v,
                                                          (short)0, acc0[s], false, false);
        acc1[s] = __builtin_amdgcn_wmma_f32_16x16x32_bf16(false, Al.v, false, Bh1.v,
                                                          (short)0, acc1[s], false, false);
      }
    }

    const float c2A = c2[codeA];
    const float c2B = c2[codeB];
#pragma unroll
    for (int s = 0; s < 4; ++s) {
#pragma unroll
      for (int j = 0; j < 8; ++j) {
        float candA = c2A - 2.0f * acc0[s][j];
        if (candA < mind[s][j]) { mind[s][j] = candA; mini[s][j] = codeA; }
        float candB = c2B - 2.0f * acc1[s][j];
        if (candB < mind[s][j]) { mind[s][j] = candB; mini[s][j] = codeB; }
      }
    }
  }

  // Cross-lane argmin within each 16-lane half (wave32 shuffles)
#pragma unroll
  for (int s = 0; s < 4; ++s) {
#pragma unroll
    for (int j = 0; j < 8; ++j) {
      float d = mind[s][j]; int bi = mini[s][j];
#pragma unroll
      for (int m = 8; m >= 1; m >>= 1) {
        float od = __shfl_xor(d, m, 32);
        int   oi = __shfl_xor(bi, m, 32);
        if (od < d || (od == d && oi < bi)) { d = od; bi = oi; }
      }
      if (col == 0) {
        int lrow = s * 16 + hf * 8 + j;    // matches C-fragment row mapping
        redd[wave * MT + lrow] = d;
        redi[wave * MT + lrow] = bi;
      }
    }
  }
  __syncthreads();

  // Cross-wave argmin, write final index (int for gather, float for output)
  if (tid < MT) {
    float d = redd[tid]; int bi = redi[tid];
#pragma unroll
    for (int w = 1; w < 8; ++w) {
      float od = redd[w * MT + tid]; int oi = redi[w * MT + tid];
      if (od < d || (od == d && oi < bi)) { d = od; bi = oi; }
    }
    idx_out[row_base + tid]  = bi;
    idxf_out[row_base + tid] = (float)bi;
  }
}

// ---------------------------------------------------------------------------
// Gather z_q (== z_q_st forward value), per-block loss partials, histogram
// ---------------------------------------------------------------------------
__global__ void vq_gather(const float* __restrict__ z_e,
                          const float* __restrict__ cb,
                          const int* __restrict__ idx,
                          float* __restrict__ zq,
                          int* __restrict__ hist,
                          float* __restrict__ partial) {
  __shared__ float red[256];
  const int tid = threadIdx.x;
  const int rb = blockIdx.x * 16;
  float local = 0.f;
  for (int r = 0; r < 16; ++r) {
    const int row = rb + r;
    const int code = idx[row];
    const float* src = cb  + (size_t)code * CODE_DIM;
    const float* zr  = z_e + (size_t)row  * CODE_DIM;
    float* dq        = zq  + (size_t)row  * CODE_DIM;
    for (int e = tid; e < CODE_DIM; e += 256) {
      float q = src[e];
      float x = zr[e];
      dq[e] = q;
      float df = q - x;
      local += df * df;
    }
  }
  if (tid < 16) atomicAdd(&hist[idx[rb + tid]], 1);
  red[tid] = local;
  __syncthreads();
  for (int off = 128; off > 0; off >>= 1) {
    if (tid < off) red[tid] += red[tid + off];
    __syncthreads();
  }
  if (tid == 0) partial[blockIdx.x] = red[0];
}

// ---------------------------------------------------------------------------
// Losses + perplexity
// ---------------------------------------------------------------------------
__global__ void vq_finalize(const int* __restrict__ hist,
                            const float* __restrict__ partial,
                            float* __restrict__ out3) {
  __shared__ float red[256];
  const int tid = threadIdx.x;

  float ls = 0.f;
  for (int k = tid; k < NROWS / 16; k += 256) ls += partial[k];
  red[tid] = ls;
  __syncthreads();
  for (int off = 128; off > 0; off >>= 1) {
    if (tid < off) red[tid] += red[tid + off];
    __syncthreads();
  }
  float loss = red[0] / (float)((size_t)NROWS * CODE_DIM);
  __syncthreads();

  float h = 0.f;
  for (int k = tid; k < NUM_CODES; k += 256) {
    float p = (float)hist[k] * (1.0f / (float)NROWS);
    h += p * logf(p + 1e-10f);            // p==0 contributes ~0, matches ref
  }
  red[tid] = h;
  __syncthreads();
  for (int off = 128; off > 0; off >>= 1) {
    if (tid < off) red[tid] += red[tid + off];
    __syncthreads();
  }
  if (tid == 0) {
    out3[0] = loss;          // commitment_loss
    out3[1] = loss;          // codebook_loss (same forward value)
    out3[2] = expf(-red[0]); // perplexity
  }
}

// ---------------------------------------------------------------------------
extern "C" void kernel_launch(void* const* d_in, const int* in_sizes, int n_in,
                              void* d_out, int out_size, void* d_ws, size_t ws_size,
                              hipStream_t stream) {
  const float* z_e = (const float*)d_in[0];
  const float* cb  = (const float*)d_in[1];

  char* ws = (char*)d_ws;
  unsigned short* chi   = (unsigned short*)(ws);             //  8,388,608 B
  unsigned short* clo   = (unsigned short*)(ws +  8388608);  //  8,388,608 B
  float*          c2    = (float*)(ws + 16777216);           //     32,768 B
  int*            idx   = (int*)  (ws + 16809984);           //     65,536 B
  int*            hist  = (int*)  (ws + 16875520);           //     32,768 B
  float*          part  = (float*)(ws + 16908288);           //      4,096 B

  float* out_zq  = (float*)d_out;
  float* out_idx = out_zq + (size_t)NROWS * CODE_DIM;
  float* out_sc  = out_idx + NROWS;

  vq_zero<<<(NUM_CODES + 255) / 256, 256, 0, stream>>>(hist);
  vq_prep_codebook<<<NUM_CODES, 256, 0, stream>>>(cb, chi, clo, c2);

  const size_t lds_bytes = (size_t)2 * MT * LDS_ROW * sizeof(unsigned short)
                         + (size_t)8 * MT * (sizeof(float) + sizeof(int)); // 137,216
  vq_argmin<<<NROWS / MT, 256, lds_bytes, stream>>>(z_e, chi, clo, c2, idx, out_idx);

  vq_gather<<<NROWS / 16, 256, 0, stream>>>(z_e, cb, idx, out_zq, hist, part);
  vq_finalize<<<1, 256, 0, stream>>>(hist, part, out_sc);
}
